// QuesAnsByRGCN_50130858279579
// MI455X (gfx1250) — matrix operand order repaired and checked by reading, not compiled, not run on hardware
//
#include <hip/hip_runtime.h>
#include <hip/hip_bf16.h>

#define N_NODES   20000
#define N_EDGES   320000
#define N_RELS    200
#define DESC_LEN  16
#define DIM_G     64
#define DIM_WORD  300
#define DIM_HID   256
#define H_DIR     128
#define N_CLASS   2000
#define Bq        4
#define Tq        32

typedef __attribute__((ext_vector_type(16))) _Float16 v16h;
typedef __attribute__((ext_vector_type(8)))  float    v8f;

// ---- workspace layout (float offsets) ----
#define OFF_GXF   ((size_t)0)                    // fwd gx: B*T*384
#define OFF_GXB   ((size_t)49152)
#define OFF_QEMB  ((size_t)98304)                // [4,256]
#define OFF_QG    ((size_t)99328)                // [4,64]
#define OFF_NF    ((size_t)99584)                // [4,N,64]
#define OFF_AGG   ((size_t)5219584)              // [4,N,64]
#define OFF_F2    ((size_t)10339584)             // [4,N,64]
#define OFF_SC2   ((size_t)15459584)             // [4,N]
#define OFF_HID   ((size_t)15539584)             // [4,1024]
#define OFF_NAGG  ((size_t)15543680)             // [4,64]
#define OFF_BF16  ((size_t)15543936)             // bases in WMMA frag layout: 4096 halves (2048 floats)

__device__ __forceinline__ float sigmoidf_(float x) { return 1.0f / (1.0f + expf(-x)); }

// ---------------- K1: gx = x @ Wx + bx for both directions ----------------
__global__ void k_gx(const int* __restrict__ questions,
                     const float* __restrict__ emb_word,
                     const float* __restrict__ Wx_f, const float* __restrict__ bx_f,
                     const float* __restrict__ Wx_b, const float* __restrict__ bx_b,
                     float* __restrict__ ws) {
    __shared__ float xrow[304];
    int blk = blockIdx.x;
    int dir = blk / (Bq * Tq);
    int row = blk % (Bq * Tq);
    int b = row / Tq, t = row % Tq;
    int tt = dir ? (Tq - 1 - t) : t;
    int word = questions[b * Tq + tt];
    const float* xr = emb_word + (size_t)word * DIM_WORD;
    int tid = threadIdx.x;
    if (tid < DIM_WORD) xrow[tid] = xr[tid];
    __syncthreads();
    const float* Wx = dir ? Wx_b : Wx_f;
    const float* bx = dir ? bx_b : bx_f;
    float acc = bx[tid];
    for (int k = 0; k < DIM_WORD; ++k)
        acc = fmaf(xrow[k], Wx[(size_t)k * 384 + tid], acc);
    float* gx = ws + (dir ? OFF_GXB : OFF_GXF);
    gx[(size_t)row * 384 + tid] = acc;
}

// ---------------- K2: GRU recurrence (2 blocks: fwd/bwd) ----------------
__global__ void k_gru(const int* __restrict__ questions,
                      const float* __restrict__ Wh_f, const float* __restrict__ bh_f,
                      const float* __restrict__ Wh_b, const float* __restrict__ bh_b,
                      float* __restrict__ ws) {
    __shared__ float h[Bq][H_DIR];
    __shared__ float gh[Bq][384];
    int dir = blockIdx.x;
    int j = threadIdx.x;           // 0..383
    const float* Wh = dir ? Wh_b : Wh_f;
    const float* bh = dir ? bh_b : bh_f;
    const float* gx = ws + (dir ? OFF_GXB : OFF_GXF);
    if (j < H_DIR) for (int b = 0; b < Bq; ++b) h[b][j] = 0.0f;
    __syncthreads();
    for (int t = 0; t < Tq; ++t) {
        for (int b = 0; b < Bq; ++b) {
            float acc = bh[j];
            for (int k = 0; k < H_DIR; ++k)
                acc = fmaf(h[b][k], Wh[(size_t)k * 384 + j], acc);
            gh[b][j] = acc;
        }
        __syncthreads();
        if (j < H_DIR) {
            for (int b = 0; b < Bq; ++b) {
                size_t gbase = ((size_t)b * Tq + t) * 384;
                float r = sigmoidf_(gx[gbase + j]       + gh[b][j]);
                float z = sigmoidf_(gx[gbase + 128 + j] + gh[b][128 + j]);
                float n = tanhf   (gx[gbase + 256 + j] + r * gh[b][256 + j]);
                float hn = (1.0f - z) * n + z * h[b][j];
                int tt = dir ? (Tq - 1 - t) : t;
                bool m = questions[b * Tq + tt] != 0;
                h[b][j] = m ? hn : h[b][j];
            }
        }
        __syncthreads();
    }
    if (j < H_DIR)
        for (int b = 0; b < Bq; ++b)
            ws[OFF_QEMB + (size_t)b * DIM_HID + dir * H_DIR + j] = h[b][j];
}

// ---------------- K3: q_g = q_emb @ W_hg + b_hg ----------------
__global__ void k_qg(const float* __restrict__ W_hg, const float* __restrict__ b_hg,
                     float* __restrict__ ws) {
    int t = threadIdx.x;            // 256 = 4 * 64
    int b = t / DIM_G, j = t % DIM_G;
    const float* qe = ws + OFF_QEMB + (size_t)b * DIM_HID;
    float acc = b_hg[j];
    for (int k = 0; k < DIM_HID; ++k)
        acc = fmaf(qe[k], W_hg[(size_t)k * DIM_G + j], acc);
    ws[OFF_QG + (size_t)b * DIM_G + j] = acc;
}

// ---------------- K3b: pre-swizzle bases into WMMA B-fragment f16 layout ----------------
// fragment group = (kt, nt, lane); 16 halves per lane, contiguous in memory.
__global__ void k_prep_bases(const float* __restrict__ bases, float* __restrict__ ws) {
    int tid = threadIdx.x;           // 256 threads = 2kt * 4nt * 32lane
    int kt = tid >> 7;
    int nt = (tid >> 5) & 3;
    int lane = tid & 31;
    int laneHi = lane >> 4, lm = lane & 15;
    _Float16* dst = (_Float16*)(ws + OFF_BF16) + (size_t)tid * 16;
#pragma unroll
    for (int j = 0; j < 16; ++j) {
        int k = kt * 32 + (j & 7) + (laneHi ? 8 : 0) + (j >= 8 ? 16 : 0);
        dst[j] = (_Float16)bases[(size_t)k * DIM_G + nt * 16 + lm];
    }
}

// ---------------- K4: fused desc gather + attention + weighted sum ----------------
__global__ void k_descattn(const int* __restrict__ node_descs,
                           const float* __restrict__ emb_desc,
                           float* __restrict__ ws) {
    __shared__ float desc[4][DESC_LEN][DIM_G + 1];
    __shared__ float qg[Bq][DIM_G];
    __shared__ float sc[4][Bq][DESC_LEN];
    int tid = threadIdx.x;
    int slot = tid >> 6, g = tid & 63;
    if (tid < Bq * DIM_G) qg[tid / DIM_G][tid % DIM_G] = ws[OFF_QG + tid];
    int n = blockIdx.x * 4 + slot;
    for (int l = 0; l < DESC_LEN; ++l) {
        int idx = node_descs[(size_t)n * DESC_LEN + l];
        desc[slot][l][g] = emb_desc[(size_t)idx * DIM_G + g];
    }
    __syncthreads();
    {
        int l = g >> 2, b = g & 3;
        float s = 0.0f;
        for (int k = 0; k < DIM_G; ++k) s = fmaf(desc[slot][l][k], qg[b][k], s);
        sc[slot][b][l] = s;
    }
    __syncthreads();
    if (g < Bq) {
        int b = g;
        float mx = sc[slot][b][0];
        for (int l = 1; l < DESC_LEN; ++l) mx = fmaxf(mx, sc[slot][b][l]);
        float sum = 0.0f;
        for (int l = 0; l < DESC_LEN; ++l) { float e = expf(sc[slot][b][l] - mx); sc[slot][b][l] = e; sum += e; }
        float inv = 1.0f / sum;
        for (int l = 0; l < DESC_LEN; ++l) sc[slot][b][l] *= inv;
    }
    __syncthreads();
    for (int b = 0; b < Bq; ++b) {
        float acc = 0.0f;
        for (int l = 0; l < DESC_LEN; ++l)
            acc = fmaf(sc[slot][b][l], desc[slot][l][g], acc);
        ws[OFF_NF + ((size_t)b * N_NODES + n) * DIM_G + g] = acc;
    }
}

// ---------------- K5: edge scatter ----------------
__global__ void k_scatter(const int* __restrict__ edge_src, const int* __restrict__ edge_dst,
                          const int* __restrict__ edge_type, const float* __restrict__ w_comp,
                          float* __restrict__ ws) {
    size_t idx = (size_t)blockIdx.x * blockDim.x + threadIdx.x;   // E*64 exactly
    int g = (int)(idx & 63);
    int e = (int)(idx >> 6);
    int s = edge_src[e], d = edge_dst[e];
    float w = w_comp[edge_type[e]];
    const float* nf = ws + OFF_NF;
    float* agg = ws + OFF_AGG;
    for (int b = 0; b < Bq; ++b) {
        float v = w * nf[((size_t)b * N_NODES + s) * DIM_G + g];
        atomicAdd(&agg[((size_t)b * N_NODES + d) * DIM_G + g], v);
    }
}

// ---------------- K6: WMMA GEMM  feat2 = relu(agg @ bases + bias) ----------------
// rows = B*N = 80000, K = 64, N' = 64.  256 threads = 8 waves, 16 rows/wave.
// B fragments read pre-swizzled f16 from global (8KB, L0/L2 resident); no LDS.
__global__ void k_rgcn_gemm(const float* __restrict__ rgcn_bias, float* __restrict__ ws) {
    int tid = threadIdx.x;
    int wave = tid >> 5, lane = tid & 31;
    int laneHi = lane >> 4;
    int lm = lane & 15;
    size_t row0 = ((size_t)blockIdx.x * 8 + wave) * 16;
    const float* A = ws + OFF_AGG + row0 * DIM_G;
    float* D = ws + OFF_F2 + row0 * DIM_G;
    const v16h* bfrag = (const v16h*)(ws + OFF_BF16);

    v8f acc[4];
#pragma unroll
    for (int nt = 0; nt < 4; ++nt)
#pragma unroll
        for (int v = 0; v < 8; ++v) acc[nt][v] = 0.0f;

    const float4* Ar = (const float4*)(A + (size_t)lm * DIM_G);
    const int off = laneHi ? 8 : 0;

#pragma unroll
    for (int kt = 0; kt < 2; ++kt) {
        const int kk = kt * 32;
        // A fragment: interleaved 16x32 layout = 4 contiguous float4 runs per lane
        float4 a0 = Ar[(kk + off) >> 2];
        float4 a1 = Ar[((kk + off) >> 2) + 1];
        float4 a2 = Ar[(kk + off + 16) >> 2];
        float4 a3 = Ar[((kk + off + 16) >> 2) + 1];
        v16h a;
        a[0] = (_Float16)a0.x;  a[1] = (_Float16)a0.y;  a[2] = (_Float16)a0.z;  a[3] = (_Float16)a0.w;
        a[4] = (_Float16)a1.x;  a[5] = (_Float16)a1.y;  a[6] = (_Float16)a1.z;  a[7] = (_Float16)a1.w;
        a[8] = (_Float16)a2.x;  a[9] = (_Float16)a2.y;  a[10] = (_Float16)a2.z; a[11] = (_Float16)a2.w;
        a[12] = (_Float16)a3.x; a[13] = (_Float16)a3.y; a[14] = (_Float16)a3.z; a[15] = (_Float16)a3.w;
#pragma unroll
        for (int nt = 0; nt < 4; ++nt) {
            v16h bf = bfrag[(kt * 4 + nt) * 32 + lane];   // one 32B contiguous load
            acc[nt] = __builtin_amdgcn_wmma_f32_16x16x32_f16(
                false, a, false, bf, (short)0, acc[nt], false, false);
        }
    }
#pragma unroll
    for (int nt = 0; nt < 4; ++nt) {
        int col = nt * 16 + lm;
        float bc = rgcn_bias[col];
#pragma unroll
        for (int v = 0; v < 8; ++v) {
            int r = v + laneHi * 8;
            float val = acc[nt][v] + bc;
            D[(size_t)r * DIM_G + col] = val > 0.0f ? val : 0.0f;
        }
    }
}

// ---------------- K7: attn2 scores = feat2 . q_g ----------------
__global__ void k_scores(float* __restrict__ ws) {
    size_t idx = (size_t)blockIdx.x * blockDim.x + threadIdx.x;
    if (idx >= (size_t)Bq * N_NODES) return;
    int b = (int)(idx / N_NODES);
    const float4* f = (const float4*)(ws + OFF_F2 + idx * DIM_G);
    const float4* q = (const float4*)(ws + OFF_QG + (size_t)b * DIM_G);
    float s = 0.0f;
#pragma unroll
    for (int k = 0; k < DIM_G / 4; ++k) {
        float4 a = f[k], c = q[k];
        s = fmaf(a.x, c.x, s); s = fmaf(a.y, c.y, s);
        s = fmaf(a.z, c.z, s); s = fmaf(a.w, c.w, s);
    }
    ws[OFF_SC2 + idx] = s;
}

// ---------------- K8: softmax over nodes + pooled node_agg (1 block / batch) ----------------
__global__ void k_pool(float* __restrict__ ws) {
    __shared__ float red[1024];
    __shared__ float sMax, sSum;
    int b = blockIdx.x, tid = threadIdx.x;
    const float* sc = ws + OFF_SC2 + (size_t)b * N_NODES;
    const float* f2 = ws + OFF_F2 + (size_t)b * N_NODES * DIM_G;
    float lm = -3.0e38f;
    for (int n = tid; n < N_NODES; n += 1024) lm = fmaxf(lm, sc[n]);
    red[tid] = lm; __syncthreads();
    for (int s = 512; s > 0; s >>= 1) { if (tid < s) red[tid] = fmaxf(red[tid], red[tid + s]); __syncthreads(); }
    if (tid == 0) sMax = red[0];
    __syncthreads();
    float ls = 0.0f;
    for (int n = tid; n < N_NODES; n += 1024) ls += expf(sc[n] - sMax);
    red[tid] = ls; __syncthreads();
    for (int s = 512; s > 0; s >>= 1) { if (tid < s) red[tid] += red[tid + s]; __syncthreads(); }
    if (tid == 0) sSum = red[0];
    __syncthreads();
    int g = tid & 63;
    float acc = 0.0f;
    for (int n = tid >> 6; n < N_NODES; n += 16) {
        float w = expf(sc[n] - sMax);
        acc = fmaf(w, f2[(size_t)n * DIM_G + g], acc);
    }
    red[tid] = acc; __syncthreads();
    for (int s = 512; s >= 64; s >>= 1) { if (tid < s) red[tid] += red[tid + s]; __syncthreads(); }
    if (tid < DIM_G)
        ws[OFF_NAGG + (size_t)b * DIM_G + tid] = red[tid] / sSum;
}

// ---------------- K9a: hidden = relu([node_agg|q_emb] @ W1 + b1) ----------------
__global__ void k_fc1(const float* __restrict__ W1, const float* __restrict__ b1,
                      float* __restrict__ ws) {
    __shared__ float feat[DIM_G + DIM_HID];
    int b = blockIdx.x, j = threadIdx.x;   // 1024 threads
    if (j < DIM_G) feat[j] = ws[OFF_NAGG + (size_t)b * DIM_G + j];
    else if (j < DIM_G + DIM_HID) feat[j] = ws[OFF_QEMB + (size_t)b * DIM_HID + (j - DIM_G)];
    __syncthreads();
    float acc = b1[j];
    for (int k = 0; k < DIM_G + DIM_HID; ++k)
        acc = fmaf(feat[k], W1[(size_t)k * 1024 + j], acc);
    ws[OFF_HID + (size_t)b * 1024 + j] = acc > 0.0f ? acc : 0.0f;
}

// ---------------- K9b: logits = hidden @ W2 + b2 ----------------
__global__ void k_fc2(const float* __restrict__ W2, const float* __restrict__ b2,
                      const float* __restrict__ ws, float* __restrict__ out) {
    size_t idx = (size_t)blockIdx.x * blockDim.x + threadIdx.x;
    if (idx >= (size_t)Bq * N_CLASS) return;
    int b = (int)(idx / N_CLASS), j = (int)(idx % N_CLASS);
    const float* h = ws + OFF_HID + (size_t)b * 1024;
    float acc = b2[j];
    for (int k = 0; k < 1024; ++k)
        acc = fmaf(h[k], W2[(size_t)k * N_CLASS + j], acc);
    out[idx] = acc;
}

extern "C" void kernel_launch(void* const* d_in, const int* in_sizes, int n_in,
                              void* d_out, int out_size, void* d_ws, size_t ws_size,
                              hipStream_t stream) {
    const int*   questions = (const int*)d_in[0];
    const int*   node_descs = (const int*)d_in[1];
    const int*   edge_src  = (const int*)d_in[2];
    const int*   edge_dst  = (const int*)d_in[3];
    const int*   edge_type = (const int*)d_in[4];
    const float* emb_word  = (const float*)d_in[5];
    const float* emb_desc  = (const float*)d_in[6];
    const float* Wx_f = (const float*)d_in[7];
    const float* Wh_f = (const float*)d_in[8];
    const float* bx_f = (const float*)d_in[9];
    const float* bh_f = (const float*)d_in[10];
    const float* Wx_b = (const float*)d_in[11];
    const float* Wh_b = (const float*)d_in[12];
    const float* bx_b = (const float*)d_in[13];
    const float* bh_b = (const float*)d_in[14];
    const float* W_hg = (const float*)d_in[15];
    const float* b_hg = (const float*)d_in[16];
    const float* bases = (const float*)d_in[17];
    const float* w_comp = (const float*)d_in[18];
    const float* rgcn_bias = (const float*)d_in[19];
    const float* W1 = (const float*)d_in[20];
    const float* b1 = (const float*)d_in[21];
    const float* W2 = (const float*)d_in[22];
    const float* b2 = (const float*)d_in[23];
    float* ws = (float*)d_ws;
    float* out = (float*)d_out;

    k_gx<<<2 * Bq * Tq, 384, 0, stream>>>(questions, emb_word, Wx_f, bx_f, Wx_b, bx_b, ws);
    k_gru<<<2, 384, 0, stream>>>(questions, Wh_f, bh_f, Wh_b, bh_b, ws);
    k_qg<<<1, 256, 0, stream>>>(W_hg, b_hg, ws);
    k_prep_bases<<<1, 256, 0, stream>>>(bases, ws);
    k_descattn<<<N_NODES / 4, 256, 0, stream>>>(node_descs, emb_desc, ws);
    hipMemsetAsync(ws + OFF_AGG, 0, (size_t)Bq * N_NODES * DIM_G * sizeof(float), stream);
    k_scatter<<<(N_EDGES * 64) / 256, 256, 0, stream>>>(edge_src, edge_dst, edge_type, w_comp, ws);
    k_rgcn_gemm<<<(Bq * N_NODES) / 128, 256, 0, stream>>>(rgcn_bias, ws);
    k_scores<<<(Bq * N_NODES + 255) / 256, 256, 0, stream>>>(ws);
    k_pool<<<Bq, 1024, 0, stream>>>(ws);
    k_fc1<<<Bq, 1024, 0, stream>>>(W1, b1, ws);
    k_fc2<<<(Bq * N_CLASS + 255) / 256, 256, 0, stream>>>(W2, b2, ws, out);
}
